// Attention_85495618994941
// MI455X (gfx1250) — compile-verified, matrix operand order checked by reading, not run on hardware
//
#include <hip/hip_runtime.h>
#include <hip/hip_bf16.h>

typedef __attribute__((ext_vector_type(16))) __bf16 v16bf;
typedef __attribute__((ext_vector_type(8)))  __bf16 v8bf;
typedef __attribute__((ext_vector_type(2)))  __bf16 v2bf;
typedef __attribute__((ext_vector_type(8)))  float  v8f;
typedef __attribute__((ext_vector_type(4)))  unsigned int v4u;
typedef __attribute__((ext_vector_type(8)))  int    v8i_;
typedef __attribute__((ext_vector_type(4)))  int    v4i_;

static constexpr int BB = 32;    // batch
static constexpr int TE = 2048;  // encoder length
static constexpr int DD = 512;   // enc dim
static constexpr int TD = 128;   // decode steps
static constexpr int UU = 512;   // lstm units
static constexpr int TCH = 8;    // t-chunks for context partial reduction

#if defined(__has_builtin)
#  if __has_builtin(__builtin_amdgcn_tensor_load_to_lds) && \
      __has_builtin(__builtin_amdgcn_s_wait_tensorcnt)
#    define HAVE_TDM 1
#  endif
#endif
#ifndef HAVE_TDM
#  define HAVE_TDM 0
#endif

// ---------------------------------------------------------------------------
// fp32 -> bf16 elementwise convert (4 elems / thread)
// ---------------------------------------------------------------------------
__global__ void cvt_bf16(const float* __restrict__ in, __bf16* __restrict__ out, long n) {
    long i = ((long)blockIdx.x * blockDim.x + threadIdx.x) * 4;
    if (i + 3 < n) {
        float4 v = *(const float4*)(in + i);
        out[i + 0] = (__bf16)v.x;
        out[i + 1] = (__bf16)v.y;
        out[i + 2] = (__bf16)v.z;
        out[i + 3] = (__bf16)v.w;
    } else {
        for (long j = i; j < n; ++j) out[j] = (__bf16)in[j];
    }
}

// fp32 [K,N] row-major  ->  bf16 [N,K] row-major (transpose)
__global__ void cvt_transpose_bf16(const float* __restrict__ in, __bf16* __restrict__ out,
                                   int K, int N) {
    long i = (long)blockIdx.x * blockDim.x + threadIdx.x;
    if (i < (long)K * N) {
        int k = (int)(i / N);
        int n = (int)(i - (long)k * N);
        out[(long)n * K + k] = (__bf16)in[i];
    }
}

__global__ void init_state(float* h, float* c, __bf16* hb) {
    int i = blockIdx.x * blockDim.x + threadIdx.x;
    if (i < BB * UU) { h[i] = 0.f; c[i] = 0.f; hb[i] = (__bf16)0.f; }
}

// ---------------------------------------------------------------------------
// WMMA bf16 GEMM:  C[M,N] = A1[M,K1] @ B1^T + (A2[M,K2] @ B2^T) + bias
// One wave32 per 16x64 output tile: 4 f32 accumulators share one A fragment
// per k-step (4 wmma : 10 b128 loads). B pre-transposed [N,K] row-major so
// every fragment is two contiguous 16B loads matching the ISA A-layout
// (lane<16 -> K 0..7 & 16..23, lane>=16 -> K 8..15 & 24..31).
// C layout: VGPR r -> M = r (lanes 0-15) / 8+r (lanes 16-31), N = lane&15.
// Requires M%16==0, N%64==0, K%32==0.
// ---------------------------------------------------------------------------
__global__ void gemm_bf16_wmma(const __bf16* __restrict__ A1, long lda1,
                               const __bf16* __restrict__ B1, long ldb1,
                               const __bf16* __restrict__ A2, long lda2,
                               const __bf16* __restrict__ B2, long ldb2,
                               const float* __restrict__ bias,
                               float* __restrict__ Cf, __bf16* __restrict__ Cb,
                               long ldc, int M, int N, int K1, int K2) {
    int wave = blockIdx.x * (blockDim.x >> 5) + (threadIdx.x >> 5);
    int lane = threadIdx.x & 31;
    int tilesN = N >> 6;                 // 64-wide N tiles
    int tilesM = M >> 4;
    if (wave >= tilesM * tilesN) return;
    int tileM = wave / tilesN;
    int tileN = wave - tileM * tilesN;
    int r15 = lane & 15;
    int kb  = (lane >> 4) << 3;          // 0 or 8

    v8f acc[4] = {};

    {
        const __bf16* a = A1 + (long)(tileM * 16 + r15) * lda1;
        const __bf16* b = B1 + (long)(tileN * 64 + r15) * ldb1;
        for (int k0 = 0; k0 < K1; k0 += 32) {
            v8bf alo = *(const v8bf*)(a + k0 + kb);
            v8bf ahi = *(const v8bf*)(a + k0 + kb + 16);
            v16bf av = __builtin_shufflevector(alo, ahi,
                0,1,2,3,4,5,6,7,8,9,10,11,12,13,14,15);
            #pragma unroll
            for (int j = 0; j < 4; ++j) {
                const __bf16* bj = b + (long)(j * 16) * ldb1 + k0 + kb;
                v8bf blo = *(const v8bf*)(bj);
                v8bf bhi = *(const v8bf*)(bj + 16);
                v16bf bv = __builtin_shufflevector(blo, bhi,
                    0,1,2,3,4,5,6,7,8,9,10,11,12,13,14,15);
                acc[j] = __builtin_amdgcn_wmma_f32_16x16x32_bf16(
                    false, av, false, bv, (short)0, acc[j], false, false);
            }
        }
    }
    if (A2 != nullptr) {
        const __bf16* a = A2 + (long)(tileM * 16 + r15) * lda2;
        const __bf16* b = B2 + (long)(tileN * 64 + r15) * ldb2;
        for (int k0 = 0; k0 < K2; k0 += 32) {
            v8bf alo = *(const v8bf*)(a + k0 + kb);
            v8bf ahi = *(const v8bf*)(a + k0 + kb + 16);
            v16bf av = __builtin_shufflevector(alo, ahi,
                0,1,2,3,4,5,6,7,8,9,10,11,12,13,14,15);
            #pragma unroll
            for (int j = 0; j < 4; ++j) {
                const __bf16* bj = b + (long)(j * 16) * ldb2 + k0 + kb;
                v8bf blo = *(const v8bf*)(bj);
                v8bf bhi = *(const v8bf*)(bj + 16);
                v16bf bv = __builtin_shufflevector(blo, bhi,
                    0,1,2,3,4,5,6,7,8,9,10,11,12,13,14,15);
                acc[j] = __builtin_amdgcn_wmma_f32_16x16x32_bf16(
                    false, av, false, bv, (short)0, acc[j], false, false);
            }
        }
    }

    int mbase = tileM * 16 + kb;         // kb==8 <=> lanes 16-31 hold M=8..15
    #pragma unroll
    for (int j = 0; j < 4; ++j) {
        int n = tileN * 64 + j * 16 + r15;
        float bb = bias ? bias[n] : 0.f;
        #pragma unroll
        for (int r = 0; r < 8; ++r) {
            float v = acc[j][r] + bb;
            long idx = (long)(mbase + r) * ldc + n;
            if (Cf) Cf[idx] = v;
            if (Cb) Cb[idx] = (__bf16)v;
        }
    }
}

// ---------------------------------------------------------------------------
// scores[b,t] = sum_d V[d] * tanh(xW1[b,t,d] + hW2[b,d])
// One wave per encoder position t (8 t per block). The 8x512 bf16 xW1 tile is
// staged into LDS by the Tensor Data Mover (wave 0 issues, all waves consume
// after the barrier); hW2 row + V also staged in LDS.
// ---------------------------------------------------------------------------
__global__ void attn_scores(const __bf16* __restrict__ xW1,
                            const float* __restrict__ hW2,
                            const float* __restrict__ V,
                            float* __restrict__ scores) {
    __shared__ __bf16 sx[8 * DD];        // 8KB TDM-staged tile
    __shared__ float sh[DD];
    __shared__ float sv[DD];
    int tid  = threadIdx.x;
    int b    = blockIdx.x >> 8;          // TE/8 == 256 blocks per batch row
    int tblk = blockIdx.x & 255;
    const __bf16* gsrc = xW1 + ((long)b * TE + tblk * 8) * DD;

#if HAVE_TDM
    if (tid < 32) {
        // Build D# per CDNA5 ISA 8.3/8.4: 2D tile 512 x 8, bf16 (data_size=1),
        // tensor_dim0=512, stride0=512, tensor_dim1=8, type=2 ("image").
        unsigned long long ga = (unsigned long long)(uintptr_t)gsrc;
        unsigned int lds = (unsigned int)(uintptr_t)(&sx[0]);
        v4u g0;
        g0[0] = 1u;                                            // count=1
        g0[1] = lds;                                           // lds_addr
        g0[2] = (unsigned int)(ga & 0xffffffffu);              // global_addr lo
        g0[3] = (unsigned int)((ga >> 32) & 0x1ffffffu) | (2u << 30); // hi|type
        v8i_ g1;
        g1[0] = 0x00010000;          // data_size=1 (2 bytes), mask=0
        g1[1] = (int)(512u << 16);   // tensor_dim0[15:0] in [31:16]
        g1[2] = (int)(8u << 16);     // tensor_dim0 hi=0; tensor_dim1[15:0]=8
        g1[3] = (int)(512u << 16);   // tensor_dim1 hi=0; tile_dim0=512
        g1[4] = 8;                   // tile_dim1=8, tile_dim2=0
        g1[5] = 512;                 // tensor_dim0_stride lo32
        g1[6] = 0;
        g1[7] = 0;
        v4i_ z4 = {0, 0, 0, 0};
#if defined(__clang_major__) && (__clang_major__ >= 23)
        v8i_ z8 = {0, 0, 0, 0, 0, 0, 0, 0};
        __builtin_amdgcn_tensor_load_to_lds(g0, g1, z4, z4, z8, 0);
#else
        __builtin_amdgcn_tensor_load_to_lds(g0, g1, z4, z4, 0);
#endif
        __builtin_amdgcn_s_wait_tensorcnt(0);
    }
    for (int i = tid; i < DD; i += 256) { sh[i] = hW2[b * DD + i]; sv[i] = V[i]; }
    __syncthreads();

    int wv = tid >> 5, lane = tid & 31;
    const __bf16* xp = &sx[wv * DD + lane * 16];
#else
    for (int i = tid; i < DD; i += 256) { sh[i] = hW2[b * DD + i]; sv[i] = V[i]; }
    __syncthreads();
    int wv = tid >> 5, lane = tid & 31;
    const __bf16* xp = gsrc + (long)wv * DD + lane * 16;
#endif

    v8bf x0 = *(const v8bf*)xp;
    v8bf x1 = *(const v8bf*)(xp + 8);
    int d0 = lane * 16;
    float s = 0.f;
    #pragma unroll
    for (int j = 0; j < 8; ++j) {
        s += sv[d0 + j]     * tanhf((float)x0[j] + sh[d0 + j]);
        s += sv[d0 + 8 + j] * tanhf((float)x1[j] + sh[d0 + 8 + j]);
    }
    #pragma unroll
    for (int off = 16; off > 0; off >>= 1) s += __shfl_xor(s, off, 32);
    if (lane == 0) scores[(long)b * TE + tblk * 8 + wv] = s;
}

// softmax over T_enc, one block per batch row
__global__ void softmax_rows(const float* __restrict__ scores, float* __restrict__ probs) {
    __shared__ float red[256];
    int b = blockIdx.x, tid = threadIdx.x;
    const float* s = scores + (long)b * TE;
    float* p = probs + (long)b * TE;

    float m = -3.4e38f;
    for (int t = tid; t < TE; t += 256) m = fmaxf(m, s[t]);
    red[tid] = m; __syncthreads();
    for (int w = 128; w > 0; w >>= 1) {
        if (tid < w) red[tid] = fmaxf(red[tid], red[tid + w]);
        __syncthreads();
    }
    float mx = red[0];
    __syncthreads();

    float sum = 0.f;
    for (int t = tid; t < TE; t += 256) { float e = __expf(s[t] - mx); p[t] = e; sum += e; }
    red[tid] = sum; __syncthreads();
    for (int w = 128; w > 0; w >>= 1) {
        if (tid < w) red[tid] += red[tid + w];
        __syncthreads();
    }
    float inv = 1.f / red[0];
    for (int t = tid; t < TE; t += 256) p[t] *= inv;
}

// Xa partials: grid (BB*TCH); block handles 256 t's for all 512 e.
// Deterministic two-stage reduction (no float atomics).
__global__ void attn_context_part(const float* __restrict__ probs,
                                  const __bf16* __restrict__ enc,
                                  float* __restrict__ part) {
    __shared__ float sa[TE / TCH];
    int b = blockIdx.x / TCH, ch = blockIdx.x % TCH;
    int tid = threadIdx.x;
    int t0 = ch * (TE / TCH);
    for (int i = tid; i < TE / TCH; i += 256) sa[i] = probs[(long)b * TE + t0 + i];
    __syncthreads();
    int e = tid * 2;
    const __bf16* ep = enc + ((long)b * TE + t0) * DD + e;
    float a0 = 0.f, a1 = 0.f;
    for (int t = 0; t < TE / TCH; ++t) {
        v2bf v = *(const v2bf*)(ep + (long)t * DD);
        a0 += sa[t] * (float)v[0];
        a1 += sa[t] * (float)v[1];
    }
    float* pp = part + ((long)ch * BB + b) * DD + e;
    pp[0] = a0;
    pp[1] = a1;
}

__global__ void attn_context_reduce(const float* __restrict__ part,
                                    __bf16* __restrict__ Xa) {
    int idx = blockIdx.x * blockDim.x + threadIdx.x;   // 0 .. BB*DD-1
    float s = 0.f;
    #pragma unroll
    for (int ch = 0; ch < TCH; ++ch) s += part[(long)ch * BB * DD + idx];
    Xa[idx] = (__bf16)s;
}

// LSTM pointwise: gates from z[32, 4*U], update c/h, emit y[:, tstep, :]
__global__ void lstm_gates(const float* __restrict__ z, float* __restrict__ c,
                           float* __restrict__ h, __bf16* __restrict__ hb,
                           float* __restrict__ y, int tstep) {
    int idx = blockIdx.x * blockDim.x + threadIdx.x;   // 0 .. BB*UU-1
    int b = idx >> 9, u = idx & 511;
    const float* zr = z + (long)b * (4 * UU);
    float gi = 1.f / (1.f + __expf(-zr[u]));
    float gf = 1.f / (1.f + __expf(-zr[UU + u]));
    float gg = tanhf(zr[2 * UU + u]);
    float go = 1.f / (1.f + __expf(-zr[3 * UU + u]));
    float cn = gf * c[idx] + gi * gg;
    float hn = go * tanhf(cn);
    c[idx] = cn;
    h[idx] = hn;
    hb[idx] = (__bf16)hn;
    y[((long)b * TD + tstep) * UU + u] = hn;
}

// ---------------------------------------------------------------------------
extern "C" void kernel_launch(void* const* d_in, const int* in_sizes, int n_in,
                              void* d_out, int out_size, void* d_ws, size_t ws_size,
                              hipStream_t stream) {
    const float* enc = (const float*)d_in[0];   // [32,2048,512]
    const float* dec = (const float*)d_in[1];   // [32,128,512]
    const float* W1  = (const float*)d_in[2];   // [512,512]
    const float* W2  = (const float*)d_in[3];   // [512,512]
    const float* b2  = (const float*)d_in[4];   // [512]
    const float* V   = (const float*)d_in[5];   // [512]
    const float* W3  = (const float*)d_in[6];   // [1024,512]
    const float* b3  = (const float*)d_in[7];   // [512]
    const float* LK  = (const float*)d_in[8];   // [512,2048]
    const float* LR  = (const float*)d_in[9];   // [512,2048]
    const float* Lb  = (const float*)d_in[10];  // [2048]
    float* y = (float*)d_out;                   // [32,128,512]

    // bf16 copies of both streamed tensors (64+64 MB) keep the per-step
    // working set inside the 192 MB L2; transposed bf16 weights for WMMA.
    char* p = (char*)d_ws;
    auto carve = [&](size_t bytes) -> void* {
        void* r = (void*)p;
        p += (bytes + 255) & ~(size_t)255;
        return r;
    };
    __bf16* enc_bf  = (__bf16*)carve((size_t)BB * TE * DD * 2);
    __bf16* xW1_bf  = (__bf16*)carve((size_t)BB * TE * DD * 2);
    __bf16* dec_bf  = (__bf16*)carve((size_t)BB * TD * UU * 2);
    __bf16* W1T     = (__bf16*)carve((size_t)DD * DD * 2);          // [512,512]
    __bf16* W2T     = (__bf16*)carve((size_t)DD * UU * 2);          // [512,512]
    __bf16* W3T     = (__bf16*)carve((size_t)UU * (DD + UU) * 2);   // [512,1024]
    __bf16* KT      = (__bf16*)carve((size_t)(4 * UU) * UU * 2);    // [2048,512]
    __bf16* RT      = (__bf16*)carve((size_t)(4 * UU) * UU * 2);    // [2048,512]
    float*  hW2f    = (float*)carve((size_t)BB * DD * 4);
    float*  hst     = (float*)carve((size_t)BB * UU * 4);
    float*  cst     = (float*)carve((size_t)BB * UU * 4);
    __bf16* hb      = (__bf16*)carve((size_t)BB * UU * 2);
    float*  scores  = (float*)carve((size_t)BB * TE * 4);
    float*  probs   = (float*)carve((size_t)BB * TE * 4);
    float*  ctxpart = (float*)carve((size_t)TCH * BB * DD * 4);
    __bf16* Xa_bf   = (__bf16*)carve((size_t)BB * DD * 2);
    __bf16* hin_bf  = (__bf16*)carve((size_t)BB * UU * 2);
    float*  zbuf    = (float*)carve((size_t)BB * 4 * UU * 4);
    (void)ws_size; (void)in_sizes; (void)n_in; (void)out_size;

    auto gemm = [&](const __bf16* A1, long lda1, const __bf16* B1, long ldb1,
                    const __bf16* A2, long lda2, const __bf16* B2, long ldb2,
                    const float* bias, float* Cf, __bf16* Cb, long ldc,
                    int M, int N, int K1, int K2) {
        int waves = (M >> 4) * (N >> 6);
        int blocks = (waves + 7) / 8;
        gemm_bf16_wmma<<<blocks, 256, 0, stream>>>(A1, lda1, B1, ldb1, A2, lda2, B2, ldb2,
                                                   bias, Cf, Cb, ldc, M, N, K1, K2);
    };

    // ---- one-time setup ---------------------------------------------------
    long nEnc = (long)BB * TE * DD;
    long nDec = (long)BB * TD * UU;
    cvt_bf16<<<(int)((nEnc / 4 + 255) / 256), 256, 0, stream>>>(enc, enc_bf, nEnc);
    cvt_bf16<<<(int)((nDec / 4 + 255) / 256), 256, 0, stream>>>(dec, dec_bf, nDec);
    cvt_transpose_bf16<<<(DD * DD + 255) / 256, 256, 0, stream>>>(W1, W1T, DD, DD);
    cvt_transpose_bf16<<<(DD * DD + 255) / 256, 256, 0, stream>>>(W2, W2T, UU, DD);
    cvt_transpose_bf16<<<((DD + UU) * UU + 255) / 256, 256, 0, stream>>>(W3, W3T, DD + UU, UU);
    cvt_transpose_bf16<<<(UU * 4 * UU + 255) / 256, 256, 0, stream>>>(LK, KT, UU, 4 * UU);
    cvt_transpose_bf16<<<(UU * 4 * UU + 255) / 256, 256, 0, stream>>>(LR, RT, UU, 4 * UU);
    init_state<<<(BB * UU + 255) / 256, 256, 0, stream>>>(hst, cst, hb);

    // xW1 = enc @ W1  (M=65536, N=512, K=512) -> bf16
    gemm(enc_bf, DD, W1T, DD, nullptr, 0, nullptr, 0,
         nullptr, nullptr, xW1_bf, DD, BB * TE, DD, DD, 0);

    // ---- sequential decode ------------------------------------------------
    for (int t = 0; t < TD; ++t) {
        // hW2 = h @ W2 + b2                     [32,512]
        gemm(hb, UU, W2T, UU, nullptr, 0, nullptr, 0,
             b2, hW2f, nullptr, DD, BB, DD, UU, 0);
        // scores / softmax / context
        attn_scores<<<BB * TE / 8, 256, 0, stream>>>(xW1_bf, hW2f, V, scores);
        softmax_rows<<<BB, 256, 0, stream>>>(scores, probs);
        attn_context_part<<<BB * TCH, 256, 0, stream>>>(probs, enc_bf, ctxpart);
        attn_context_reduce<<<(BB * DD + 255) / 256, 256, 0, stream>>>(ctxpart, Xa_bf);
        // h_in = [x_t, Xa] @ W3 + b3           [32,512]
        gemm(dec_bf + (long)t * UU, (long)TD * UU, W3T, DD + UU,
             Xa_bf, DD, W3T + DD, DD + UU,
             b3, nullptr, hin_bf, UU, BB, UU, UU, DD);
        // z = h_in @ K + h @ R + bias          [32,2048]
        gemm(hin_bf, UU, KT, UU, hb, UU, RT, UU,
             Lb, zbuf, nullptr, 4 * UU, BB, 4 * UU, UU, UU);
        // gates + state update + output row
        lstm_gates<<<(BB * UU + 255) / 256, 256, 0, stream>>>(zbuf, cst, hst, hb, y, t);
    }
}
